// Gemma2Attention_66941360275492
// MI455X (gfx1250) — matrix operand order repaired and checked by reading, not compiled, not run on hardware
//
#include <hip/hip_runtime.h>

// ---------------------------------------------------------------------------
// Gemma2 attention block for gfx1250 (MI455X), bf16 WMMA pipeline.
//   K0 : f32 -> bf16 convert (hidden) ; f32 -> bf16 TRANSPOSED convert (weights)
//   K1 : tiled WMMA GEMM, double-buffered GLOBAL_LOAD_ASYNC_TO_LDS staging
//   K2 : RoPE + repack to [B][H][S][HD] bf16
//   K3 : V transpose to [B][KV][HD][S] bf16
//   K4 : flash attention (tanh softcap, causal, online softmax), bf16 WMMA
// ---------------------------------------------------------------------------

#define B_  2
#define S_  2048
#define H_  3584
#define NH_ 16
#define HD_ 256
#define NKV_ 8
#define QKD (NH_*HD_)    // 4096
#define KVD (NKV_*HD_)   // 2048

typedef __attribute__((ext_vector_type(16))) __bf16 v16bf;
typedef __attribute__((ext_vector_type(8)))  float  v8f;
typedef int v4i __attribute__((vector_size(16)));   // matches builtin pointee type

union Frag {            // one wave32 WMMA 16-bit operand: 16 bf16 per lane
    v16bf v;
    uint4 q[2];
};

__device__ __forceinline__ __bf16 f2bf(float x) {
    union { float f; unsigned u; } v; v.f = x;
    unsigned r = v.u + 0x7FFFu + ((v.u >> 16) & 1u);   // round-to-nearest-even
    union { unsigned short s; __bf16 b; } o;
    o.s = (unsigned short)(r >> 16);
    return o.b;
}

__device__ __forceinline__ v8f wmma_bf16(const Frag& a, const Frag& b, v8f c) {
    return __builtin_amdgcn_wmma_f32_16x16x32_bf16(false, a.v, false, b.v,
                                                   (short)0, c, false, false);
}

// ---- CDNA5 async copy: global -> LDS (ASYNCcnt-tracked), 16B per lane -----
#if __has_builtin(__builtin_amdgcn_global_load_async_to_lds_b128)
#define HAVE_ASYNC_BUILTIN 1
#endif

__device__ __forceinline__ void async_copy16(const __bf16* g, __bf16* l) {
#if defined(HAVE_ASYNC_BUILTIN)
    // param0: AS1 (global) v4i*  [from clang diagnostic]; param1: LDS dest
    __builtin_amdgcn_global_load_async_to_lds_b128(
        (__attribute__((address_space(1))) v4i*)g,
        (__attribute__((address_space(3))) v4i*)l, 0, 0);
#else
    // inline-asm fallback: VDST = LDS byte offset (low 32 bits of generic
    // shared address), VADDR = 64-bit global address
    unsigned lofs = (unsigned)(unsigned long long)l;
    asm volatile("global_load_async_to_lds_b128 %0, %1, off"
                 :: "v"(lofs), "v"(g) : "memory");
#endif
}

__device__ __forceinline__ void wait_async_le4() {
#if __has_builtin(__builtin_amdgcn_s_wait_asynccnt)
    __builtin_amdgcn_s_wait_asynccnt(4);
#else
    asm volatile("s_wait_asynccnt 0x4" ::: "memory");
#endif
}
__device__ __forceinline__ void wait_async_0() {
#if __has_builtin(__builtin_amdgcn_s_wait_asynccnt)
    __builtin_amdgcn_s_wait_asynccnt(0);
#else
    asm volatile("s_wait_asynccnt 0x0" ::: "memory");
#endif
}

// ---------------------------------------------------------------- K0: convert
__global__ void cvt_f32_bf16(const float* __restrict__ src,
                             __bf16* __restrict__ dst, long n) {
    long i = (long)blockIdx.x * blockDim.x + threadIdx.x;
    if (i < n) dst[i] = f2bf(src[i]);
}

// f32 [K][N] -> bf16 transposed [N][K]; indexed over the destination so the
// stores coalesce (strided reads hit L2, one-time cost).
__global__ void cvt_transpose_f32_bf16(const float* __restrict__ src,
                                       __bf16* __restrict__ dst, int N, int K) {
    long i = (long)blockIdx.x * blockDim.x + threadIdx.x;
    int k = (int)(i % K);
    int n = (int)(i / K);
    dst[i] = f2bf(src[(long)k * N + n]);
}

// ------------------------------------------------------------- K1: WMMA GEMM
// C[4096,N] f32 = A[4096,K] bf16 (row-major) * Bt[N,K] bf16 (pre-transposed).
// 128x128 block, BK=32, 256 threads (8 waves, 2x4 wave grid, 64x32 per wave).
// Double-buffered: tile k+1 streams into LDS via GLOBAL_LOAD_ASYNC_TO_LDS_B128
// while tile k feeds the WMMA pipe.  ld = 40 bf16 (80B): 16B aligned and
// conflict-free for ds_load_b128 fragment reads.
template <int N, int K>
__global__ __launch_bounds__(256)
void gemm_bf16_wmma(const __bf16* __restrict__ A, const __bf16* __restrict__ Bt,
                    float* __restrict__ C) {
    __shared__ __bf16 As[2][128 * 40];
    __shared__ __bf16 Bs[2][128 * 40];

    const int tid = threadIdx.x;
    const int wid = tid >> 5, lane = tid & 31;
    const int lo = lane & 15, hi = lane >> 4;
    const int wm = wid >> 2, wn = wid & 3;           // 2 x 4 wave grid
    const int m0 = blockIdx.y * 128, n0 = blockIdx.x * 128;

    // both tiles are 128 rows x 32 K of bf16: 512 16B chunks, 2 per thread each
    auto issue = [&](int buf, int kt) {
        #pragma unroll
        for (int it = 0; it < 2; ++it) {
            int c = it * 256 + tid;
            int r = c >> 2, kc = (c & 3) * 8;
            async_copy16(A  + (long)(m0 + r) * K + kt + kc, &As[buf][r * 40 + kc]);
            async_copy16(Bt + (long)(n0 + r) * K + kt + kc, &Bs[buf][r * 40 + kc]);
        }
    };

    v8f acc[4][2];
    #pragma unroll
    for (int i = 0; i < 4; ++i)
        #pragma unroll
        for (int j = 0; j < 2; ++j) acc[i][j] = (v8f){0,0,0,0,0,0,0,0};

    constexpr int NK = K / 32;
    issue(0, 0);

    #pragma unroll 1
    for (int i = 0; i < NK; ++i) {
        if (i + 1 < NK) {
            issue((i + 1) & 1, (i + 1) * 32);
            wait_async_le4();                        // previous tile landed
        } else {
            wait_async_0();
        }
        __syncthreads();

        const __bf16* as = As[i & 1];
        const __bf16* bs = Bs[i & 1];
        Frag a[4], b[2];
        #pragma unroll
        for (int mi = 0; mi < 4; ++mi) {             // A frag: row lo, K chunks hi*8, 16+hi*8
            int r = wm * 64 + mi * 16 + lo;
            a[mi].q[0] = *(const uint4*)(&as[r * 40 + hi * 8]);
            a[mi].q[1] = *(const uint4*)(&as[r * 40 + 16 + hi * 8]);
        }
        #pragma unroll
        for (int ni = 0; ni < 2; ++ni) {             // B frag: col lo, K chunks hi*8, 16+hi*8
            int n = wn * 32 + ni * 16 + lo;
            b[ni].q[0] = *(const uint4*)(&bs[n * 40 + hi * 8]);
            b[ni].q[1] = *(const uint4*)(&bs[n * 40 + 16 + hi * 8]);
        }
        #pragma unroll
        for (int mi = 0; mi < 4; ++mi)
            #pragma unroll
            for (int ni = 0; ni < 2; ++ni)
                acc[mi][ni] = wmma_bf16(a[mi], b[ni], acc[mi][ni]);
        __syncthreads();                             // tile consumed; safe to overwrite
    }

    #pragma unroll
    for (int mi = 0; mi < 4; ++mi)
        #pragma unroll
        for (int ni = 0; ni < 2; ++ni) {
            const int col = n0 + wn * 32 + ni * 16 + lo;
            float* cp = C + (long)(m0 + wm * 64 + mi * 16 + hi * 8) * N + col;
            #pragma unroll
            for (int e = 0; e < 8; ++e)              // C layout: row = e + 8*hi, col = lo
                cp[(long)e * N] = acc[mi][ni][e];
        }
}

// ------------------------------------------------------- K2: RoPE + repack
// src: [B*S][nh*HD] f32 (token-major GEMM output); dst: [B][nh][S][HD] bf16
__global__ void rope_pack(const float* __restrict__ src, __bf16* __restrict__ dst,
                          int nh) {
    long i = (long)blockIdx.x * blockDim.x + threadIdx.x;
    int d = (int)(i % HD_);
    long t = i / HD_;
    int s = (int)(t % S_); t /= S_;
    int h = (int)(t % nh);
    int b = (int)(t / nh);
    int i2 = d & 127;
    // inv_freq = 10000^(-i2/128)
    float inv = __expf(-(float)i2 * (9.210340371976184f / 128.f));
    float ang = (float)s * inv;
    float sn, cs;
    __sincosf(ang, &sn, &cs);
    const float* row = src + (long)(b * S_ + s) * (nh * HD_) + h * HD_;
    float x  = row[d];
    float xo = row[(d < 128) ? d + 128 : d - 128];
    float r  = (d < 128) ? (x * cs - xo * sn) : (x * cs + xo * sn);
    dst[(((long)(b * nh + h)) * S_ + s) * HD_ + d] = f2bf(r);
}

// --------------------------------------------------------- K3: V transpose
// src: [B*S][KVD] f32 ; dst: [B][NKV][HD][S] bf16
__global__ void v_transpose(const float* __restrict__ src, __bf16* __restrict__ dst) {
    long i = (long)blockIdx.x * blockDim.x + threadIdx.x;
    int d = (int)(i % HD_);
    long t = i / HD_;
    int s = (int)(t % S_); t /= S_;
    int kv = (int)(t % NKV_);
    int b  = (int)(t / NKV_);
    float x = src[(long)(b * S_ + s) * KVD + kv * HD_ + d];
    dst[(((long)(b * NKV_ + kv)) * HD_ + d) * S_ + s] = f2bf(x);
}

// ------------------------------------------------------ K4: flash attention
// Q: [B][NH][S][HD] bf16 ; K: [B][NKV][S][HD] bf16 ; VT: [B][NKV][HD][S] bf16
// Out: [B*S][NH*HD] bf16 (token-major for the output projection)
// Block = (qt 16-row tile, head, batch); 4 waves; each wave owns 64 HD columns.
__global__ __launch_bounds__(128)
void flash_attn(const __bf16* __restrict__ Q, const __bf16* __restrict__ Km,
                const __bf16* __restrict__ VT, __bf16* __restrict__ Out) {
    const float SCALE = 0.0625f;                  // 256^-0.5
    const int qt = blockIdx.x, h = blockIdx.y, b = blockIdx.z;
    const int kvh = h >> 1;                       // GROUPS = NH/NKV = 2
    const int tid = threadIdx.x;
    const int wid = tid >> 5, lane = tid & 31;
    const int lo = lane & 15, hi = lane >> 4;

    __shared__ __bf16 Pl[4][16 * 40];             // per-wave P tile [16][32], ld=40

    const __bf16* qp = Q  + (((long)(b * NH_ + h)) * S_ + qt * 16) * HD_;
    const __bf16* kp = Km + ((long)(b * NKV_ + kvh)) * S_ * HD_;
    const __bf16* vp = VT + ((long)(b * NKV_ + kvh)) * HD_ * S_;

    // Q fragments: 16x256 as 8 A-frags (16x32 each)
    Frag qa[8];
    #pragma unroll
    for (int kf = 0; kf < 8; ++kf) {
        const __bf16* rp = qp + lo * HD_ + kf * 32;
        qa[kf].q[0] = *(const uint4*)(rp + hi * 8);
        qa[kf].q[1] = *(const uint4*)(rp + 16 + hi * 8);
    }

    v8f o[4];
    #pragma unroll
    for (int i = 0; i < 4; ++i) o[i] = (v8f){0,0,0,0,0,0,0,0};
    float mrow[8], lrow[8];
    #pragma unroll
    for (int e = 0; e < 8; ++e) { mrow[e] = -1e30f; lrow[e] = 0.f; }

    const int ntiles = (qt >> 1) + 1;             // 32-key tiles up to the diagonal
    for (int tb = 0; tb < ntiles; ++tb) {
        const int t32 = tb * 32;
        v8f s0 = (v8f){0,0,0,0,0,0,0,0};
        v8f s1 = (v8f){0,0,0,0,0,0,0,0};
        #pragma unroll
        for (int kf = 0; kf < 8; ++kf) {          // S = Q * K^T over HD=256
            Frag bk0, bk1;
            const __bf16* r0 = kp + (long)(t32 + lo) * HD_ + kf * 32 + hi * 8;
            bk0.q[0] = *(const uint4*)(r0);
            bk0.q[1] = *(const uint4*)(r0 + 16);
            const __bf16* r1 = kp + (long)(t32 + 16 + lo) * HD_ + kf * 32 + hi * 8;
            bk1.q[0] = *(const uint4*)(r1);
            bk1.q[1] = *(const uint4*)(r1 + 16);
            s0 = wmma_bf16(qa[kf], bk0, s0);
            s1 = wmma_bf16(qa[kf], bk1, s1);
        }

        // softcap + causal mask + online softmax (C layout: row=e+8*hi, col=lo)
        #pragma unroll
        for (int e = 0; e < 8; ++e) {
            const int row = qt * 16 + e + hi * 8;
            float a0 = s0[e] * SCALE; a0 = 50.f * tanhf(a0 * 0.02f);
            float a1 = s1[e] * SCALE; a1 = 50.f * tanhf(a1 * 0.02f);
            if (t32 + lo > row)      a0 = -1e30f;
            if (t32 + 16 + lo > row) a1 = -1e30f;
            float rm = fmaxf(a0, a1);
            rm = fmaxf(rm, __shfl_xor(rm, 1, 32));
            rm = fmaxf(rm, __shfl_xor(rm, 2, 32));
            rm = fmaxf(rm, __shfl_xor(rm, 4, 32));
            rm = fmaxf(rm, __shfl_xor(rm, 8, 32));
            float mn = fmaxf(mrow[e], rm);
            float sc = __expf(mrow[e] - mn);
            float p0 = __expf(a0 - mn);
            float p1 = __expf(a1 - mn);
            float rs = p0 + p1;
            rs += __shfl_xor(rs, 1, 32);
            rs += __shfl_xor(rs, 2, 32);
            rs += __shfl_xor(rs, 4, 32);
            rs += __shfl_xor(rs, 8, 32);
            lrow[e] = lrow[e] * sc + rs;
            mrow[e] = mn;
            o[0][e] *= sc; o[1][e] *= sc; o[2][e] *= sc; o[3][e] *= sc;
            const int pr = e + hi * 8;
            Pl[wid][pr * 40 + lo]      = f2bf(p0);
            Pl[wid][pr * 40 + 16 + lo] = f2bf(p1);
        }

        // cross-lane P handoff through LDS (same wave; DS is in-order per wave)
        asm volatile("s_wait_dscnt 0" ::: "memory");
        Frag pa;
        pa.q[0] = *(const uint4*)(&Pl[wid][lo * 40 + hi * 8]);
        pa.q[1] = *(const uint4*)(&Pl[wid][lo * 40 + 16 + hi * 8]);

        // O += P * V  (B frag from VT: col = hd, K = kv positions, chunks hi*8, 16+hi*8)
        #pragma unroll
        for (int ni = 0; ni < 4; ++ni) {
            const int hd = wid * 64 + ni * 16 + lo;
            const __bf16* vr = vp + (long)hd * S_ + t32 + hi * 8;
            Frag bv;
            bv.q[0] = *(const uint4*)(vr);
            bv.q[1] = *(const uint4*)(vr + 16);
            o[ni] = wmma_bf16(pa, bv, o[ni]);
        }
    }

    // finalize: O /= l, store token-major bf16 for the output projection
    #pragma unroll
    for (int e = 0; e < 8; ++e) {
        float inv = 1.f / lrow[e];
        const int srow = qt * 16 + e + hi * 8;
        long obase = (long)(b * S_ + srow) * QKD + h * HD_;
        #pragma unroll
        for (int ni = 0; ni < 4; ++ni)
            Out[obase + wid * 64 + ni * 16 + lo] = f2bf(o[ni][e] * inv);
    }
}

// ---------------------------------------------------------------------------
extern "C" void kernel_launch(void* const* d_in, const int* in_sizes, int n_in,
                              void* d_out, int out_size, void* d_ws, size_t ws_size,
                              hipStream_t stream) {
    const float* hidden = (const float*)d_in[0];
    // d_in[1] = attention_mask: analytically causal, recomputed in-kernel
    const float* Wq = (const float*)d_in[2];
    const float* Wk = (const float*)d_in[3];
    const float* Wv = (const float*)d_in[4];
    const float* Wo = (const float*)d_in[5];
    float* out = (float*)d_out;

    const long T = (long)B_ * S_;   // 4096 tokens

    // workspace carve-up (bf16 staging + f32 GEMM outputs)
    char* p = (char*)d_ws;
    __bf16* hid16 = (__bf16*)p;            p += T * H_ * 2;
    __bf16* wqT   = (__bf16*)p;            p += (long)H_ * QKD * 2;   // [QKD][H]
    __bf16* wkT   = (__bf16*)p;            p += (long)H_ * KVD * 2;   // [KVD][H]
    __bf16* wvT   = (__bf16*)p;            p += (long)H_ * KVD * 2;   // [KVD][H]
    __bf16* woT   = (__bf16*)p;            p += (long)QKD * H_ * 2;   // [H][QKD]
    float*  qf    = (float*)p;             p += T * QKD * 4;
    float*  kf    = (float*)p;             p += T * KVD * 4;
    float*  vf    = (float*)p;             p += T * KVD * 4;
    __bf16* q16   = (__bf16*)p;            p += T * QKD * 2;
    __bf16* k16   = (__bf16*)p;            p += T * KVD * 2;
    __bf16* vt16  = (__bf16*)p;            p += T * KVD * 2;
    __bf16* att16 = (__bf16*)p;            p += T * QKD * 2;

    cvt_f32_bf16<<<dim3((unsigned)(T * H_ / 256)), 256, 0, stream>>>(hidden, hid16, T * H_);
    cvt_transpose_f32_bf16<<<dim3((unsigned)((long)H_ * QKD / 256)), 256, 0, stream>>>(Wq, wqT, QKD, H_);
    cvt_transpose_f32_bf16<<<dim3((unsigned)((long)H_ * KVD / 256)), 256, 0, stream>>>(Wk, wkT, KVD, H_);
    cvt_transpose_f32_bf16<<<dim3((unsigned)((long)H_ * KVD / 256)), 256, 0, stream>>>(Wv, wvT, KVD, H_);
    cvt_transpose_f32_bf16<<<dim3((unsigned)((long)QKD * H_ / 256)), 256, 0, stream>>>(Wo, woT, H_, QKD);

    // QKV projections: [4096 x 3584] * [3584 x N]
    gemm_bf16_wmma<QKD, H_><<<dim3(QKD / 128, T / 128), 256, 0, stream>>>(hid16, wqT, qf);
    gemm_bf16_wmma<KVD, H_><<<dim3(KVD / 128, T / 128), 256, 0, stream>>>(hid16, wkT, kf);
    gemm_bf16_wmma<KVD, H_><<<dim3(KVD / 128, T / 128), 256, 0, stream>>>(hid16, wvT, vf);

    // RoPE + head-major repack (Q, K); V transpose to [HD][S]
    rope_pack<<<dim3((unsigned)(T * QKD / 256)), 256, 0, stream>>>(qf, q16, NH_);
    rope_pack<<<dim3((unsigned)(T * KVD / 256)), 256, 0, stream>>>(kf, k16, NKV_);
    v_transpose<<<dim3((unsigned)(T * KVD / 256)), 256, 0, stream>>>(vf, vt16);

    // flash attention: one block per (16-row q tile, head, batch)
    flash_attn<<<dim3(S_ / 16, NH_, B_), 128, 0, stream>>>(q16, k16, vt16, att16);

    // output projection: [4096 x 4096] * [4096 x 3584] -> f32 d_out
    gemm_bf16_wmma<H_, QKD><<<dim3(H_ / 128, T / 128), 256, 0, stream>>>(att16, woT, out);
}
